// Encoder_28192165331217
// MI455X (gfx1250) — matrix operand order repaired and checked by reading, not compiled, not run on hardware
//
#include <hip/hip_runtime.h>
#include <hip/hip_bf16.h>

// ---------------------------------------------------------------------------
// LSTM-VAE encoder for MI455X (gfx1250), wave32 + v_wmma_f32_16x16x32_bf16.
//
// Per-step GEMM: gates[16,2048] = concat(x_t, h)[16,640] @ [W;U][640,2048]
//   - 64 blocks of 16 batch rows each (recurrence independent per row)
//   - 16 waves/block; wave w owns hidden units [32w,32w+32) across ALL 4
//     gates (8 N-tiles) -> gate nonlinearities fused in registers, cell state
//     kept in VGPRs for all 256 steps.
//   - B operand pre-repacked to fragment-native bf16 layout in d_ws (one
//     contiguous 32B load per lane per fragment -> 2x global_load_b128).
//   - A operand (x_t bf16 ++ h bf16) staged in LDS, row stride 648 halfs
//     (=324 dwords == 4 mod 64 banks -> conflict-free ds_load_b128 frags).
// ---------------------------------------------------------------------------

typedef __bf16 bf16;
typedef __attribute__((ext_vector_type(16))) __bf16 v16bf;
typedef __attribute__((ext_vector_type(8)))  __bf16 v8bf;
typedef __attribute__((ext_vector_type(4)))  __bf16 v4bf;
typedef __attribute__((ext_vector_type(8)))  float  v8f;

#define B_   1024
#define T_   256
#define D_   128
#define H_   512
#define Z_   64
#define G4H  2048          // 4*H
#define NTH  512           // threads per block (16 waves)
#define AROW 648           // LDS A row stride in halfs (1296B, 16B aligned)
#define KT   20            // 640 / 32 K-tiles
#define NTILES 128         // 2048 / 16 N-tiles

__device__ __forceinline__ float softplusf(float v) {
    // numerically stable softplus
    return fmaxf(v, 0.0f) + log1pf(__expf(-fabsf(v)));
}
__device__ __forceinline__ float sigmoidf(float v) {
    return 1.0f / (1.0f + __expf(-v));
}

// ---------------------------------------------------------------------------
// One-time repack of [W(128x2048); U(512x2048)] f32 -> bf16 B-fragment layout.
// Fragment tile (kt,nt): 512 halfs; lane l holds 16 halfs at +l*16, where
// element e corresponds to Bsrc[kt*32 + (e&7) + 8*(l>>4) + 16*(e>>3)][nt*16 + (l&15)].
// ---------------------------------------------------------------------------
__global__ void repack_weights_k(const float* __restrict__ W,
                                 const float* __restrict__ U,
                                 bf16* __restrict__ Bp) {
    int idx = blockIdx.x * 256 + threadIdx.x;
    if (idx >= KT * NTILES * 512) return;
    int e   = idx & 15;
    int l   = (idx >> 4) & 31;
    int ntk = idx >> 9;            // kt*128 + nt
    int nt  = ntk & (NTILES - 1);
    int kt  = ntk >> 7;
    int K   = kt * 32 + (e & 7) + 8 * (l >> 4) + 16 * (e >> 3);
    int col = nt * 16 + (l & 15);
    float v = (K < D_) ? W[K * G4H + col] : U[(K - D_) * G4H + col];
    Bp[(size_t)ntk * 512 + l * 16 + e] = (bf16)v;
}

// ---------------------------------------------------------------------------
// Main fused kernel: LSTM over T steps + mu/logvar/z heads.
// ---------------------------------------------------------------------------
__global__ __launch_bounds__(NTH)
void lstm_vae_k(const float* __restrict__ x,     // [B,T,D]
                const float* __restrict__ eps,   // [B,Z]
                const float* __restrict__ bias,  // [4H] (f-gate bias already 1)
                const bf16*  __restrict__ Bp,    // packed [kt][nt][512] bf16
                const float* __restrict__ Wm, const float* __restrict__ bm,
                const float* __restrict__ Wv, const float* __restrict__ bv,
                float* __restrict__ out) {
    __shared__ alignas(16) bf16 As[16 * AROW];   // [16 rows][648] (x | h | pad)

    const int tid  = threadIdx.x;
    const int lane = tid & 31;
    const int w    = tid >> 5;       // wave id 0..15
    const int rb   = blockIdx.x * 16;
    const int lr   = lane & 15;
    const int lh   = lane >> 4;

    // zero the h region (K = 128..639)
    for (int i = tid; i < 16 * H_; i += NTH) {
        int m = i >> 9, k = i & (H_ - 1);
        As[m * AROW + D_ + k] = (bf16)0.0f;
    }

    // per-lane gate biases for this wave's 32 hidden units (2 sub-tiles)
    float bg[4][2];
#pragma unroll
    for (int g = 0; g < 4; ++g)
#pragma unroll
        for (int u = 0; u < 2; ++u)
            bg[g][u] = bias[g * H_ + w * 32 + u * 16 + lr];

    // cell state in registers, C-fragment layout: (u, r) -> m=r+8*lh, n=lr+16u
    float cst[2][8];
#pragma unroll
    for (int u = 0; u < 2; ++u)
#pragma unroll
        for (int r = 0; r < 8; ++r) cst[u][r] = 0.0f;

    for (int t = 0; t < T_; ++t) {
        // ---- stage x_t -> A region K 0..127 (f32 -> bf16), coalesced ----
        {
            int m  = tid >> 5;             // 0..15
            int d0 = (tid & 31) * 4;       // 0..124
            float4 v = *(const float4*)(x + ((size_t)(rb + m) * T_ + t) * D_ + d0);
            v4bf p;
            p[0] = (bf16)v.x; p[1] = (bf16)v.y; p[2] = (bf16)v.z; p[3] = (bf16)v.w;
            *(v4bf*)(&As[m * AROW + d0]) = p;
        }
        __syncthreads();   // x staged, h(t-1) writes from prev iter complete

        // ---- GEMM: acc[8] = A(16x640) @ B(640x 8*16-cols-of-this-wave) ----
        v8f acc[8];
#pragma unroll
        for (int i = 0; i < 8; ++i) acc[i] = (v8f){0.f,0.f,0.f,0.f,0.f,0.f,0.f,0.f};

        for (int kt = 0; kt < KT; ++kt) {
            // A fragment: lane holds A[lr][kbase+8lh .. +7] and [.. +16+8lh ..]
            const bf16* ar = &As[lr * AROW + kt * 32 + 8 * lh];
            v16bf aF;
            ((v8bf*)&aF)[0] = *(const v8bf*)(ar);
            ((v8bf*)&aF)[1] = *(const v8bf*)(ar + 16);
#pragma unroll
            for (int nt = 0; nt < 8; ++nt) {
                const int g4 = nt >> 1, u = nt & 1;
                const int ntile = g4 * 32 + w * 2 + u;     // global N-tile
                const v16bf bF = *(const v16bf*)(
                    Bp + ((size_t)(kt * NTILES + ntile) * 512) + lane * 16);
                acc[nt] = __builtin_amdgcn_wmma_f32_16x16x32_bf16(
                    false, aF, false, bF, (short)0, acc[nt], false, false);
            }
        }
        __syncthreads();   // all waves done reading A before h overwrite

        // ---- fused gates + cell update + h writeback (registers only) ----
#pragma unroll
        for (int u = 0; u < 2; ++u) {
            const int hcol = D_ + w * 32 + u * 16 + lr;   // A column of h[n]
#pragma unroll
            for (int r = 0; r < 8; ++r) {
                const int m = r + 8 * lh;
                float iv = sigmoidf(acc[0 + u][r] + bg[0][u]);
                float fv = sigmoidf(acc[2 + u][r] + bg[1][u]);
                float gv = softplusf(acc[4 + u][r] + bg[2][u]);
                float ov = sigmoidf(acc[6 + u][r] + bg[3][u]);
                float cn = fv * cst[u][r] + iv * gv;
                cst[u][r] = cn;
                As[m * AROW + hcol] = (bf16)(ov * softplusf(cn));
            }
        }
        // (barrier at top of next iteration orders h writes vs next GEMM)
    }
    __syncthreads();   // final h visible to all threads

    // ---- heads: mu = h@Wm + bm, logvar = h@Wv + bv, z = mu + eps*exp(lv/2)
    {
        const int m  = tid >> 5;       // batch row within tile
        const int nn = tid & 31;       // covers cols nn and nn+32
        const bf16* hrow = &As[m * AROW + D_];
        float m0 = 0.f, m1 = 0.f, v0 = 0.f, v1 = 0.f;
        for (int k = 0; k < H_; ++k) {
            float hv = (float)hrow[k];
            const float* wmr = Wm + k * Z_;
            const float* wvr = Wv + k * Z_;
            m0 += hv * wmr[nn];      m1 += hv * wmr[nn + 32];
            v0 += hv * wvr[nn];      v1 += hv * wvr[nn + 32];
        }
        const int row = rb + m;
        float mu0 = m0 + bm[nn],      lv0 = v0 + bv[nn];
        float mu1 = m1 + bm[nn + 32], lv1 = v1 + bv[nn + 32];
        float z0 = mu0 + eps[row * Z_ + nn]      * __expf(0.5f * lv0);
        float z1 = mu1 + eps[row * Z_ + nn + 32] * __expf(0.5f * lv1);
        out[              row * Z_ + nn]      = mu0;
        out[              row * Z_ + nn + 32] = mu1;
        out[B_ * Z_     + row * Z_ + nn]      = lv0;
        out[B_ * Z_     + row * Z_ + nn + 32] = lv1;
        out[2 * B_ * Z_ + row * Z_ + nn]      = z0;
        out[2 * B_ * Z_ + row * Z_ + nn + 32] = z1;
    }
}

// ---------------------------------------------------------------------------
extern "C" void kernel_launch(void* const* d_in, const int* in_sizes, int n_in,
                              void* d_out, int out_size, void* d_ws, size_t ws_size,
                              hipStream_t stream) {
    const float* x   = (const float*)d_in[0];
    const float* eps = (const float*)d_in[1];
    const float* W   = (const float*)d_in[2];
    const float* U   = (const float*)d_in[3];
    const float* b   = (const float*)d_in[4];
    const float* Wm  = (const float*)d_in[5];
    const float* bm  = (const float*)d_in[6];
    const float* Wv  = (const float*)d_in[7];
    const float* bv  = (const float*)d_in[8];
    float* out = (float*)d_out;

    // packed bf16 weights: 20*128*512 halfs = 2.5 MiB in d_ws
    bf16* Bp = (bf16*)d_ws;
    const int total = KT * NTILES * 512;
    repack_weights_k<<<(total + 255) / 256, 256, 0, stream>>>(W, U, Bp);
    lstm_vae_k<<<B_ / 16, NTH, 0, stream>>>(x, eps, b, Bp, Wm, bm, Wv, bv, out);
}
